// MHA_39230231281877
// MI455X (gfx1250) — compile-verified, matrix operand order checked by reading, not run on hardware
//
#include <hip/hip_runtime.h>
#include <hip/hip_bf16.h>

// Problem constants (match reference)
#define BB  2
#define SS  2048
#define DD  1024
#define HH  16
#define DHH 64

typedef __attribute__((ext_vector_type(16))) __bf16 v16bf;
typedef __attribute__((ext_vector_type(8)))  __bf16 v8bf;
typedef __attribute__((ext_vector_type(8)))  float  v8f;
typedef __attribute__((ext_vector_type(4)))  float  v4f;

// base-2 softmax folding: attn_scale * log2(e) = 0.125 * 1.4426950408889634
#define C2SCALE 0.18033688011112042f

// Native bf16 conversion: gfx1250 has hardware bf16 cvt (backend packs pairs).
__device__ __forceinline__ __bf16 f2bf(float f) { return (__bf16)f; }

// Build a 16-element bf16 WMMA fragment from a row pointer:
// per ISA 7.12.2, lane's elements are K = kb + half*8 + {0..7} (elems 0-7)
// and K = kb + 16 + half*8 + {0..7} (elems 8-15): two contiguous 16B runs.
__device__ __forceinline__ v16bf load_frag(const __bf16* row, int kb, int half) {
    union { v16bf v; v8bf h[2]; } u;
    u.h[0] = *(const v8bf*)(row + kb + half * 8);
    u.h[1] = *(const v8bf*)(row + kb + 16 + half * 8);
    return u.v;
}

__device__ __forceinline__ v8f wmma_bf16(v16bf a, v16bf b, v8f c) {
    return __builtin_amdgcn_wmma_f32_16x16x32_bf16(false, a, false, b,
                                                   (short)0, c, false, false);
}

// ---------------------------------------------------------------------------
// Phase 1: fp32 -> bf16 conversion, 8 elements per thread (2x b128 in, 1x b128 out)
// ---------------------------------------------------------------------------
__global__ __launch_bounds__(256) void cvt_bf16(const float* __restrict__ src,
                                                __bf16* __restrict__ dst, int n8) {
    int i = blockIdx.x * blockDim.x + threadIdx.x;
    if (i >= n8) return;
    v4f a = ((const v4f*)src)[i * 2];
    v4f b = ((const v4f*)src)[i * 2 + 1];
    v8bf o;
#pragma unroll
    for (int j = 0; j < 4; ++j) { o[j] = f2bf(a[j]); o[j + 4] = f2bf(b[j]); }
    ((v8bf*)dst)[i] = o;
}

// ---------------------------------------------------------------------------
// Phase 2/4: Y = X @ W^T  (X: [B*S, D] bf16, W: [D, D] bf16 row-major)
// One wave computes a 16x64 strip of Y; K-loop is register double-buffered.
// MODE 0 (Qh/Kh [B,H,S,DH]) and MODE 3 (fp32 [B*S,D]) compute Y^T = W * X^T
// so every epilogue store is a packed b128. MODE 2 ([B,H,DH,S], V^T) keeps
// the original orientation (consecutive s packs instead).
// PRESCALE: multiply output by C2SCALE (used for Q so attention scores come
// out of the WMMA already in the base-2 softmax domain).
// ---------------------------------------------------------------------------
template <int MODE, int PRESCALE>
__global__ __launch_bounds__(256) void gemm_xwT(const __bf16* __restrict__ X,
                                                const __bf16* __restrict__ W,
                                                void* __restrict__ out) {
    const int lane = threadIdx.x & 31;
    const int wave = threadIdx.x >> 5;
    const int wg   = blockIdx.x * 8 + wave;
    const int M_TILES = (BB * SS) / 16;          // 256
    const int mtile = wg % M_TILES;
    const int ngrp  = wg / M_TILES;              // 0..15 (64 cols each)
    const int half  = lane >> 4;
    const int l15   = lane & 15;
    const int m0 = mtile * 16;
    const int n0 = ngrp * 64;

    const __bf16* xrow = X + (size_t)(m0 + l15) * DD;
    const __bf16* wrow[4];
#pragma unroll
    for (int nt = 0; nt < 4; ++nt)
        wrow[nt] = W + (size_t)(n0 + nt * 16 + l15) * DD;

    v8f acc[4] = {};

    // ---- software-pipelined K loop: prime k=0 fragments ----
    v16bf x_cur = load_frag(xrow, 0, half);
    v16bf w_cur[4];
#pragma unroll
    for (int nt = 0; nt < 4; ++nt) w_cur[nt] = load_frag(wrow[nt], 0, half);

    for (int k = 0; k < DD; k += 32) {
        const int kn = (k + 32) & (DD - 1);      // wraps to 0 on final step
        v16bf x_nxt = load_frag(xrow, kn, half);
        v16bf w_nxt[4];
#pragma unroll
        for (int nt = 0; nt < 4; ++nt) w_nxt[nt] = load_frag(wrow[nt], kn, half);

#pragma unroll
        for (int nt = 0; nt < 4; ++nt) {
            if (MODE == 2)  // Y tile:  M = m, N = n
                acc[nt] = wmma_bf16(x_cur, w_cur[nt], acc[nt]);
            else            // Y^T tile: M = n, N = m
                acc[nt] = wmma_bf16(w_cur[nt], x_cur, acc[nt]);
        }

        x_cur = x_nxt;
#pragma unroll
        for (int nt = 0; nt < 4; ++nt) w_cur[nt] = w_nxt[nt];
    }

    if (PRESCALE) {
#pragma unroll
        for (int nt = 0; nt < 4; ++nt)
#pragma unroll
            for (int r = 0; r < 8; ++r) acc[nt][r] *= C2SCALE;
    }

    // ---- packed b128 epilogues ----
    if (MODE == 3) {
        // Y^T tile: n = n0 + nt*16 + 8*half + r (consecutive), m = m0 + l15
        float* orow = (float*)out + (size_t)(m0 + l15) * DD + n0;
#pragma unroll
        for (int nt = 0; nt < 4; ++nt) {
            v4f lo, hi;
#pragma unroll
            for (int j = 0; j < 4; ++j) { lo[j] = acc[nt][j]; hi[j] = acc[nt][j + 4]; }
            *(v4f*)(orow + nt * 16 + 8 * half)     = lo;
            *(v4f*)(orow + nt * 16 + 8 * half + 4) = hi;
        }
    } else if (MODE == 0) {
        // Y^T tile -> Qh/Kh [B,H,S,DH]: dh = nt*16 + 8*half + r consecutive
        const int m = m0 + l15;
        const int b = m >> 11, s = m & (SS - 1);
        const int h = n0 >> 6;                   // one 64-col group == one head
        __bf16* orow = (__bf16*)out + (((size_t)b * HH + h) * SS + s) * DHH;
#pragma unroll
        for (int nt = 0; nt < 4; ++nt) {
            v8bf pk;
#pragma unroll
            for (int r = 0; r < 8; ++r) pk[r] = f2bf(acc[nt][r]);
            *(v8bf*)(orow + nt * 16 + 8 * half) = pk;
        }
    } else {
        // Y tile -> VhT [B,H,DH,S]: s = m0 + 8*half + r consecutive
        const int b = m0 >> 11, s0 = m0 & (SS - 1);
        const int h = n0 >> 6;
#pragma unroll
        for (int nt = 0; nt < 4; ++nt) {
            const int dh = nt * 16 + l15;
            __bf16* p = (__bf16*)out +
                (((size_t)b * HH + h) * DHH + dh) * SS + s0 + 8 * half;
            v8bf pk;
#pragma unroll
            for (int r = 0; r < 8; ++r) pk[r] = f2bf(acc[nt][r]);
            *(v8bf*)p = pk;
        }
    }
}

// ---------------------------------------------------------------------------
// Phase 3: causal flash attention, fully in the transposed domain.
// S^T = K Q^T  (M = key in elements, N = query in lanes): softmax stats are
// per-lane scalars (local max/sum + one shfl_xor(16)). O^T = V^T P^T keeps
// lanes = query, so the online correction is a scalar multiply.
// Q is pre-scaled by C2SCALE, so scores are directly exp2 arguments.
// The key-block loop is split: unmasked main loop + one masked tail block,
// so the hot path carries no v_cmp/v_cndmask work at all.
// Qh,Kh: [B,H,S,DH] bf16;  VhT: [B,H,DH,S] bf16;  Ctx out: [B*S, D] bf16.
// ---------------------------------------------------------------------------
__global__ __launch_bounds__(256) void attn_fwd(const __bf16* __restrict__ Qh,
                                                const __bf16* __restrict__ Kh,
                                                const __bf16* __restrict__ VhT,
                                                __bf16* __restrict__ Ctx) {
    __shared__ __align__(16) __bf16 pbuf[8][16 * 32];   // wave-private P^T tiles

    const int lane = threadIdx.x & 31;
    const int wave = threadIdx.x >> 5;
    const int wg   = blockIdx.x * 8 + wave;             // 0..4095
    const int qblk = wg % (SS / 16);
    const int bh   = wg / (SS / 16);                    // b*H + h
    const int half = lane >> 4;
    const int l15  = lane & 15;
    const int qbase = qblk * 16;
    const float NEG_INF = -__builtin_inff();

    const __bf16* Qb = Qh  + (size_t)bh * SS * DHH;
    const __bf16* Kb = Kh  + (size_t)bh * SS * DHH;
    const __bf16* Vb = VhT + (size_t)bh * DHH * SS;

    // Q fragments (B operand of S^T): lane = query, elements = dh runs.
    const __bf16* qrow = Qb + (size_t)(qbase + l15) * DHH;
    v16bf qf[2];
    qf[0] = load_frag(qrow, 0, half);
    qf[1] = load_frag(qrow, 32, half);

    v8f oacc[4] = {};                 // O^T: M = dh in elements, N = query lanes
    float rmax = NEG_INF, rsum = 0.f; // per-lane (query = qbase + l15)

    __bf16* pw = pbuf[wave];
    const int qglob = qbase + l15;
    const int nfull = qbase >> 5;     // fully-unmasked 32-key blocks

    // prime K fragments (A operand): lane = key-in-tile, elements = dh runs.
    v16bf kf[2][2];
#pragma unroll
    for (int t = 0; t < 2; ++t) {
        const __bf16* krow = Kb + (size_t)(t * 16 + l15) * DHH;
        kf[t][0] = load_frag(krow, 0, half);
        kf[t][1] = load_frag(krow, 32, half);
    }

    // ======================= unmasked main loop =======================
    for (int jb = 0; jb < nfull; ++jb) {
        const int kbase = jb * 32;

        v8f sc[2] = {};
#pragma unroll
        for (int t = 0; t < 2; ++t)
#pragma unroll
            for (int kb = 0; kb < 2; ++kb)
                sc[t] = wmma_bf16(kf[t][kb], qf[kb], sc[t]);

        // prefetch V^T fragments for this block + K fragments for next block
        v16bf vf[4];
#pragma unroll
        for (int dt = 0; dt < 4; ++dt)
            vf[dt] = load_frag(Vb + (size_t)(dt * 16 + l15) * SS + kbase, 0, half);
        v16bf kf_n[2][2];
#pragma unroll
        for (int t = 0; t < 2; ++t) {
            const __bf16* krow = Kb + (size_t)(kbase + 32 + t * 16 + l15) * DHH;
            kf_n[t][0] = load_frag(krow, 0, half);
            kf_n[t][1] = load_frag(krow, 32, half);
        }

        // online softmax (scores already in base-2 domain; no mask, no scale)
        float lm = NEG_INF;
#pragma unroll
        for (int t = 0; t < 2; ++t)
#pragma unroll
            for (int r = 0; r < 8; ++r) lm = fmaxf(lm, sc[t][r]);
        lm = fmaxf(lm, __shfl_xor(lm, 16, 32));

        const float mnew = fmaxf(rmax, lm);
        const float corr = __builtin_amdgcn_exp2f(rmax - mnew);
        rmax = mnew;

        float ls = 0.f;
#pragma unroll
        for (int t = 0; t < 2; ++t)
#pragma unroll
            for (int r = 0; r < 8; ++r) {
                const float p = __builtin_amdgcn_exp2f(sc[t][r] - mnew);
                sc[t][r] = p;
                ls += p;
            }
        ls += __shfl_xor(ls, 16, 32);
        rsum = rsum * corr + ls;
#pragma unroll
        for (int dt = 0; dt < 4; ++dt)
#pragma unroll
            for (int r = 0; r < 8; ++r) oacc[dt][r] *= corr;

        // stage P^T (lane owns 8 consecutive keys) -> 2x ds_store_b128
#pragma unroll
        for (int t = 0; t < 2; ++t) {
            v8bf pk;
#pragma unroll
            for (int r = 0; r < 8; ++r) pk[r] = f2bf(sc[t][r]);
            *(v8bf*)(pw + l15 * 32 + t * 16 + half * 8) = pk;
        }
        v16bf pf = load_frag(pw + l15 * 32, 0, half);   // 2x ds_load_b128

#pragma unroll
        for (int dt = 0; dt < 4; ++dt)
            oacc[dt] = wmma_bf16(vf[dt], pf, oacc[dt]);

#pragma unroll
        for (int t = 0; t < 2; ++t) {
            kf[t][0] = kf_n[t][0];
            kf[t][1] = kf_n[t][1];
        }
    }

    // ======================= masked diagonal block =======================
    {
        const int kbase = nfull * 32;   // == qbase & ~31; kf already holds it

        v8f sc[2] = {};
#pragma unroll
        for (int t = 0; t < 2; ++t)
#pragma unroll
            for (int kb = 0; kb < 2; ++kb)
                sc[t] = wmma_bf16(kf[t][kb], qf[kb], sc[t]);

        v16bf vf[4];
#pragma unroll
        for (int dt = 0; dt < 4; ++dt)
            vf[dt] = load_frag(Vb + (size_t)(dt * 16 + l15) * SS + kbase, 0, half);

        // causal mask: key = kbase + t*16 + 8*half + r  vs  query = qglob
#pragma unroll
        for (int t = 0; t < 2; ++t)
#pragma unroll
            for (int r = 0; r < 8; ++r) {
                const int key = kbase + t * 16 + half * 8 + r;
                sc[t][r] = (key > qglob) ? NEG_INF : sc[t][r];
            }

        float lm = NEG_INF;
#pragma unroll
        for (int t = 0; t < 2; ++t)
#pragma unroll
            for (int r = 0; r < 8; ++r) lm = fmaxf(lm, sc[t][r]);
        lm = fmaxf(lm, __shfl_xor(lm, 16, 32));

        const float mnew = fmaxf(rmax, lm);
        const float corr = __builtin_amdgcn_exp2f(rmax - mnew);
        rmax = mnew;

        float ls = 0.f;
#pragma unroll
        for (int t = 0; t < 2; ++t)
#pragma unroll
            for (int r = 0; r < 8; ++r) {
                const float p = __builtin_amdgcn_exp2f(sc[t][r] - mnew);
                sc[t][r] = p;
                ls += p;
            }
        ls += __shfl_xor(ls, 16, 32);
        rsum = rsum * corr + ls;
#pragma unroll
        for (int dt = 0; dt < 4; ++dt)
#pragma unroll
            for (int r = 0; r < 8; ++r) oacc[dt][r] *= corr;

#pragma unroll
        for (int t = 0; t < 2; ++t) {
            v8bf pk;
#pragma unroll
            for (int r = 0; r < 8; ++r) pk[r] = f2bf(sc[t][r]);
            *(v8bf*)(pw + l15 * 32 + t * 16 + half * 8) = pk;
        }
        v16bf pf = load_frag(pw + l15 * 32, 0, half);

#pragma unroll
        for (int dt = 0; dt < 4; ++dt)
            oacc[dt] = wmma_bf16(vf[dt], pf, oacc[dt]);
    }

    // ---- epilogue: O^T / rsum -> Ctx [B*S, D] bf16, packed b128 stores ----
    const int b = bh / HH, h = bh % HH;
    const float inv = 1.f / rsum;
    __bf16* orow = Ctx + ((size_t)b * SS + qglob) * DD + h * DHH;
#pragma unroll
    for (int dt = 0; dt < 4; ++dt) {
        v8bf pk;
#pragma unroll
        for (int r = 0; r < 8; ++r) pk[r] = f2bf(oacc[dt][r] * inv);
        *(v8bf*)(orow + dt * 16 + 8 * half) = pk;      // dh consecutive
    }
}

// ---------------------------------------------------------------------------
// Host launcher
// ---------------------------------------------------------------------------
extern "C" void kernel_launch(void* const* d_in, const int* in_sizes, int n_in,
                              void* d_out, int out_size, void* d_ws, size_t ws_size,
                              hipStream_t stream) {
    const float* q  = (const float*)d_in[0];
    const float* k  = (const float*)d_in[1];
    const float* v  = (const float*)d_in[2];
    // d_in[3] = mask (causal mask is computed analytically, ignored)
    const float* Wq = (const float*)d_in[4];
    const float* Wk = (const float*)d_in[5];
    const float* Wv = (const float*)d_in[6];
    const float* Wo = (const float*)d_in[7];
    float* out = (float*)d_out;

    const int NQ = BB * SS * DD;   // 4,194,304
    const int NW = DD * DD;        // 1,048,576

    char* ws = (char*)d_ws;
    size_t off = 0;
    __bf16* qbf = (__bf16*)(ws + off); off += (size_t)NQ * 2;
    __bf16* kbf = (__bf16*)(ws + off); off += (size_t)NQ * 2;
    __bf16* vbf = (__bf16*)(ws + off); off += (size_t)NQ * 2;
    __bf16* wqb = (__bf16*)(ws + off); off += (size_t)NW * 2;
    __bf16* wkb = (__bf16*)(ws + off); off += (size_t)NW * 2;
    __bf16* wvb = (__bf16*)(ws + off); off += (size_t)NW * 2;
    __bf16* wob = (__bf16*)(ws + off); off += (size_t)NW * 2;
    __bf16* Qh  = (__bf16*)(ws + off); off += (size_t)NQ * 2;
    __bf16* Kh  = (__bf16*)(ws + off); off += (size_t)NQ * 2;
    __bf16* VhT = (__bf16*)(ws + off); off += (size_t)NQ * 2;
    __bf16* Ctx = (__bf16*)(ws + off); off += (size_t)NQ * 2;

    // Phase 1: precision conversion (bandwidth-trivial: ~80 MB at 23.3 TB/s)
    cvt_bf16<<<(NQ / 8) / 256, 256, 0, stream>>>(q,  qbf, NQ / 8);
    cvt_bf16<<<(NQ / 8) / 256, 256, 0, stream>>>(k,  kbf, NQ / 8);
    cvt_bf16<<<(NQ / 8) / 256, 256, 0, stream>>>(v,  vbf, NQ / 8);
    cvt_bf16<<<(NW / 8) / 256, 256, 0, stream>>>(Wq, wqb, NW / 8);
    cvt_bf16<<<(NW / 8) / 256, 256, 0, stream>>>(Wk, wkb, NW / 8);
    cvt_bf16<<<(NW / 8) / 256, 256, 0, stream>>>(Wv, wvb, NW / 8);
    cvt_bf16<<<(NW / 8) / 256, 256, 0, stream>>>(Wo, wob, NW / 8);

    // Phase 2: projections. 4096 waves each = 512 blocks of 8 waves.
    // Q is pre-scaled into the base-2 softmax domain.
    gemm_xwT<0, 1><<<512, 256, 0, stream>>>(qbf, wqb, Qh);
    gemm_xwT<0, 0><<<512, 256, 0, stream>>>(kbf, wkb, Kh);
    gemm_xwT<2, 0><<<512, 256, 0, stream>>>(vbf, wvb, VhT);

    // Phase 3: causal flash attention. 2*16*128 = 4096 waves = 512 blocks.
    attn_fwd<<<512, 256, 0, stream>>>(Qh, Kh, VhT, Ctx);

    // Phase 4: output projection -> fp32 d_out.
    gemm_xwT<3, 0><<<512, 256, 0, stream>>>(Ctx, wob, out);
}
